// AUGRU_29214367547614
// MI455X (gfx1250) — matrix operand order repaired and checked by reading, not compiled, not run on hardware
//
#include <hip/hip_runtime.h>
#include <hip/hip_bf16.h>

typedef __attribute__((ext_vector_type(16))) __bf16 v16bf;
typedef __attribute__((ext_vector_type(8)))  float  v8f;

#define B_ 512
#define T_ 128
#define D_ 512

union BfVec { v16bf v; uint4 q[2]; };

__device__ __forceinline__ float fast_sigmoid(float x) {
    return 1.0f / (1.0f + __expf(-x));
}
__device__ __forceinline__ float fast_tanh(float x) {
    x = fminf(fmaxf(x, -15.0f), 15.0f);
    float e = __expf(2.0f * x);
    return (e - 1.0f) / (e + 1.0f);
}

__device__ __forceinline__ v8f wmma_bf16(const BfVec& a, const BfVec& b, v8f c) {
    return __builtin_amdgcn_wmma_f32_16x16x32_bf16(
        false, a.v, false, b.v, (short)0, c, false, false);
}

// ---------------------------------------------------------------------------
// Kernel 0a: pack weights as bf16, transposed so each WMMA B-operand lane does
// one contiguous 32B load.  WTu/WTr: [N=512][K=1024], K = [W rows | U rows].
// WTh/UTh: [N=512][K=512].
// ---------------------------------------------------------------------------
__global__ __launch_bounds__(256) void augru_prep_weights(
    const float* __restrict__ Wu, const float* __restrict__ Uu,
    const float* __restrict__ Wr, const float* __restrict__ Ur,
    const float* __restrict__ Wh, const float* __restrict__ Uh,
    __bf16* __restrict__ WTu, __bf16* __restrict__ WTr,
    __bf16* __restrict__ WTh, __bf16* __restrict__ UTh)
{
    int idx = blockIdx.x * blockDim.x + threadIdx.x;
    if (idx < 512 * 1024) {
        int n = idx >> 10, k = idx & 1023;
        float wu = (k < 512) ? Wu[k * 512 + n] : Uu[(k - 512) * 512 + n];
        float wr = (k < 512) ? Wr[k * 512 + n] : Ur[(k - 512) * 512 + n];
        WTu[idx] = (__bf16)wu;
        WTr[idx] = (__bf16)wr;
    }
    if (idx < 512 * 512) {
        int n = idx >> 9, k = idx & 511;
        WTh[idx] = (__bf16)Wh[k * 512 + n];
        UTh[idx] = (__bf16)Uh[k * 512 + n];
    }
}

// ---------------------------------------------------------------------------
// Kernel 0b: convert x to bf16 once (so the scan can async-DMA it straight
// into LDS with no in-flight conversion).  8 elements / thread.
// ---------------------------------------------------------------------------
__global__ __launch_bounds__(256) void augru_prep_x(
    const float* __restrict__ x, __bf16* __restrict__ xbf)
{
    size_t i = ((size_t)blockIdx.x * blockDim.x + threadIdx.x) * 8;
    const float4* s = (const float4*)(x + i);
    float4 a0 = s[0], a1 = s[1];
    alignas(16) __bf16 tmp[8] = {
        (__bf16)a0.x, (__bf16)a0.y, (__bf16)a0.z, (__bf16)a0.w,
        (__bf16)a1.x, (__bf16)a1.y, (__bf16)a1.z, (__bf16)a1.w };
    *(uint4*)(xbf + i) = *(const uint4*)tmp;
}

// ---------------------------------------------------------------------------
// Kernel 1: attention scores + softmax.  One block per batch row.
// ---------------------------------------------------------------------------
__global__ __launch_bounds__(256) void augru_attn(
    const float* __restrict__ x, const float* __restrict__ item,
    const float* __restrict__ Wa, float* __restrict__ attn)
{
    __shared__ float sItem[D_];
    __shared__ float sP[D_];
    __shared__ float sS[T_];
    __shared__ float sW[T_];

    const int b = blockIdx.x;
    const int tid = threadIdx.x;

    sItem[tid]       = item[(size_t)b * D_ + tid];
    sItem[tid + 256] = item[(size_t)b * D_ + tid + 256];
    __syncthreads();

    for (int d = tid; d < D_; d += 256) {
        const float4* wrow = (const float4*)(Wa + (size_t)d * D_);
        float acc = 0.f;
        #pragma unroll 4
        for (int e4 = 0; e4 < D_ / 4; ++e4) {
            float4 w = wrow[e4];
            acc += w.x * sItem[e4 * 4 + 0] + w.y * sItem[e4 * 4 + 1]
                 + w.z * sItem[e4 * 4 + 2] + w.w * sItem[e4 * 4 + 3];
        }
        sP[d] = acc;
    }
    __syncthreads();

    if (tid < T_) {
        const float4* xr = (const float4*)(x + (((size_t)b * T_) + tid) * D_);
        float acc = 0.f;
        #pragma unroll 4
        for (int e4 = 0; e4 < D_ / 4; ++e4) {
            float4 v = xr[e4];
            acc += v.x * sP[e4 * 4 + 0] + v.y * sP[e4 * 4 + 1]
                 + v.z * sP[e4 * 4 + 2] + v.w * sP[e4 * 4 + 3];
        }
        sS[tid] = acc;
    }
    __syncthreads();

    if (tid < T_) sW[tid] = sS[tid];
    __syncthreads();
    for (int s = 64; s >= 1; s >>= 1) {
        if (tid < s) sW[tid] = fmaxf(sW[tid], sW[tid + s]);
        __syncthreads();
    }
    float mx = sW[0];
    __syncthreads();
    float e = 0.f;
    if (tid < T_) { e = __expf(sS[tid] - mx); sW[tid] = e; }
    __syncthreads();
    for (int s = 64; s >= 1; s >>= 1) {
        if (tid < s) sW[tid] += sW[tid + s];
        __syncthreads();
    }
    if (tid < T_) attn[(size_t)b * T_ + tid] = e / sW[0];
}

// ---------------------------------------------------------------------------
// Kernel 2: fused AUGRU scan.  One block owns a 16-row batch slab for all
// 128 steps; h lives in LDS.  x(t+1) streams in via async global->LDS DMA
// (ASYNCcnt) double-buffered behind step t's WMMAs.  32 waves, one 16-col
// N-tile each; 96 v_wmma_f32_16x16x32_bf16 per wave per step.
// ---------------------------------------------------------------------------
__global__ __launch_bounds__(1024) void augru_scan(
    const __bf16* __restrict__ xbf, const float* __restrict__ attn,
    const __bf16* __restrict__ WTu, const __bf16* __restrict__ WTr,
    const __bf16* __restrict__ WTh, const __bf16* __restrict__ UTh,
    const float* __restrict__ p_bu, const float* __restrict__ p_br,
    const float* __restrict__ p_bh,
    float* __restrict__ out, float* __restrict__ hlast)
{
    __shared__ __align__(16) __bf16 sX[2][16][512];  // double-buffered x tile
    __shared__ __align__(16) __bf16 sHbf[16][512];   // h as WMMA A operand
    __shared__ float sH[16][512];                    // f32 master h
    __shared__ float sAttn[2][16];

    const int tid  = threadIdx.x;
    const int wave = tid >> 5;
    const int lane = tid & 31;
    const int half = lane >> 4;
    const int l16  = lane & 15;
    const int b0   = blockIdx.x * 16;
    const int ncol = wave * 16 + l16;

    const float bu_s = p_bu[ncol];
    const float br_s = p_br[ncol];
    const float bh_s = p_bh[ncol];

    const __bf16* rWu = WTu + (size_t)ncol * 1024 + half * 16;
    const __bf16* rWr = WTr + (size_t)ncol * 1024 + half * 16;
    const __bf16* rWh = WTh + (size_t)ncol * 512  + half * 16;
    const __bf16* rUh = UTh + (size_t)ncol * 512  + half * 16;

    // per-thread 16B slice of the x tile (8192 bf16 / 1024 threads)
    const int sr = (tid * 8) >> 9;
    const int sc = (tid * 8) & 511;
    const __bf16* xsrc = xbf + ((size_t)(b0 + sr) * T_) * D_ + sc;

    // h := 0
    for (int j = tid; j < 16 * 512; j += 1024) {
        int r = j >> 9, c = j & 511;
        sH[r][c] = 0.f;
        sHbf[r][c] = (__bf16)0.f;
    }

    // prologue: DMA x(0) into buffer 0
    {
        unsigned lds = (unsigned)(size_t)&sX[0][sr][sc];
        const __bf16* g = xsrc;                       // t = 0
        asm volatile("global_load_async_to_lds_b128 %0, %1, off"
                     :: "v"(lds), "v"(g) : "memory");
    }
    if (tid < 16) sAttn[0][tid] = attn[(size_t)(b0 + tid) * T_];
    asm volatile("s_wait_asynccnt 0" ::: "memory");
    __syncthreads();

    for (int t = 0; t < T_; ++t) {
        const int buf  = t & 1;
        const int nbuf = buf ^ 1;

        // ---- kick off async DMA of x(t+1) into the other buffer ----
        if (t + 1 < T_) {
            unsigned lds = (unsigned)(size_t)&sX[nbuf][sr][sc];
            const __bf16* g = xsrc + (size_t)(t + 1) * D_;
            asm volatile("global_load_async_to_lds_b128 %0, %1, off"
                         :: "v"(lds), "v"(g) : "memory");
            if (tid < 16)
                sAttn[nbuf][tid] = attn[(size_t)(b0 + tid) * T_ + t + 1];
        }

        // ---- WMMA GEMMs ----
        v8f cu = {}, cr2 = {}, cgx = {}, cgh = {};
        const __bf16* xrow = &sX[buf][l16][0];
        const __bf16* hrow = &sHbf[l16][0];

        // x half of K: feeds u, r, and x@Wh  (A loaded once per chunk)
        #pragma unroll 4
        for (int kk = 0; kk < 16; ++kk) {
            const int kb = kk * 32;
            BfVec a, bm;
            a.q[0] = *(const uint4*)(xrow + kb + half * 16);
            a.q[1] = *(const uint4*)(xrow + kb + half * 16 + 8);
            const uint4* bp = (const uint4*)(rWu + kb);
            bm.q[0] = bp[0]; bm.q[1] = bp[1];
            cu = wmma_bf16(a, bm, cu);
            bp = (const uint4*)(rWr + kb);
            bm.q[0] = bp[0]; bm.q[1] = bp[1];
            cr2 = wmma_bf16(a, bm, cr2);
            bp = (const uint4*)(rWh + kb);
            bm.q[0] = bp[0]; bm.q[1] = bp[1];
            cgx = wmma_bf16(a, bm, cgx);
        }
        // h half of K: feeds u, r (U-part rows at k>=512), and h@Uh
        #pragma unroll 4
        for (int kk = 0; kk < 16; ++kk) {
            const int kb = kk * 32;
            BfVec a, bm;
            a.q[0] = *(const uint4*)(hrow + kb + half * 16);
            a.q[1] = *(const uint4*)(hrow + kb + half * 16 + 8);
            const uint4* bp = (const uint4*)(rWu + 512 + kb);
            bm.q[0] = bp[0]; bm.q[1] = bp[1];
            cu = wmma_bf16(a, bm, cu);
            bp = (const uint4*)(rWr + 512 + kb);
            bm.q[0] = bp[0]; bm.q[1] = bp[1];
            cr2 = wmma_bf16(a, bm, cr2);
            bp = (const uint4*)(rUh + kb);
            bm.q[0] = bp[0]; bm.q[1] = bp[1];
            cgh = wmma_bf16(a, bm, cgh);
        }
        __syncthreads();   // all waves done reading sHbf/sH before mutation

        // ---- elementwise gate math + h update ----
        const bool last = (t == T_ - 1);
        #pragma unroll
        for (int r = 0; r < 8; ++r) {
            const int M = r + half * 8;           // C/D layout: M = r + 8*half
            float u  = fast_sigmoid(cu[r]  + bu_s);
            float rg = fast_sigmoid(cr2[r] + br_s);
            float hh = fast_tanh(cgx[r] + rg * cgh[r] + bh_s);
            float at = sAttn[buf][M];
            float hold = sH[M][ncol];
            float uh = at * u;
            float hnew = (1.0f - uh) * hold + uh * hh;
            sH[M][ncol] = hnew;
            sHbf[M][ncol] = (__bf16)hnew;
            out[(((size_t)(b0 + M)) * T_ + t) * D_ + ncol] = hnew;
            if (last) hlast[(size_t)(b0 + M) * D_ + ncol] = hnew;
        }
        // wave-local wait for our x(t+1) DMA; barrier makes it block-wide
        asm volatile("s_wait_asynccnt 0" ::: "memory");
        __syncthreads();
    }
}

// ---------------------------------------------------------------------------
extern "C" void kernel_launch(void* const* d_in, const int* in_sizes, int n_in,
                              void* d_out, int out_size, void* d_ws, size_t ws_size,
                              hipStream_t stream) {
    const float* x    = (const float*)d_in[0];
    const float* item = (const float*)d_in[1];
    const float* Wa   = (const float*)d_in[2];
    const float* Wu   = (const float*)d_in[3];
    const float* Uu   = (const float*)d_in[4];
    const float* bu   = (const float*)d_in[5];
    const float* Wr   = (const float*)d_in[6];
    const float* Ur   = (const float*)d_in[7];
    const float* br   = (const float*)d_in[8];
    const float* Wh   = (const float*)d_in[9];
    const float* Uh   = (const float*)d_in[10];
    const float* bh   = (const float*)d_in[11];

    char* ws = (char*)d_ws;
    float*  attn = (float*)ws;                                    // 256 KB
    __bf16* WTu  = (__bf16*)(ws + 262144);                        // 1 MB
    __bf16* WTr  = (__bf16*)(ws + 262144 + 1048576);              // 1 MB
    __bf16* WTh  = (__bf16*)(ws + 262144 + 2 * 1048576);          // 512 KB
    __bf16* UTh  = (__bf16*)(ws + 262144 + 2 * 1048576 + 524288); // 512 KB
    __bf16* xbf  = (__bf16*)(ws + 262144 + 3 * 1048576);          // 64 MB

    float* out   = (float*)d_out;
    float* hlast = out + (size_t)B_ * T_ * D_;

    augru_prep_weights<<<2048, 256, 0, stream>>>(Wu, Uu, Wr, Ur, Wh, Uh,
                                                 WTu, WTr, WTh, UTh);
    augru_prep_x<<<(B_ * T_ * D_ / 8 + 255) / 256, 256, 0, stream>>>(x, xbf);
    augru_attn<<<B_, 256, 0, stream>>>(x, item, Wa, attn);
    augru_scan<<<B_ / 16, 1024, 0, stream>>>(xbf, attn, WTu, WTr, WTh, UTh,
                                             bu, br, bh, out, hlast);
}
